// Decoder_82240033784333
// MI455X (gfx1250) — matrix operand order repaired and checked by reading, not compiled, not run on hardware
//
#include <hip/hip_runtime.h>

// Problem constants (from reference)
#define B_   64
#define H_   1024
#define V_   32000
#define T_   40
#define G4H  (4*H_)
#define SOS  1

typedef __attribute__((ext_vector_type(16))) __bf16 v16bf;
typedef __attribute__((ext_vector_type(8)))  float  v8f;
typedef __attribute__((ext_vector_type(8)))  __bf16 bf16x8;

union AFrag { v16bf v; bf16x8 h[2]; };

__device__ __forceinline__ float sigmoidf_(float x) { return 1.0f / (1.0f + __expf(-x)); }

// ---------------------------------------------------------------------------
// f32 -> bf16 weight conversion (one-time; halves L2 footprint: 165MB->82MB,
// keeps all weights L2-resident across the 40-step recurrence)
// ---------------------------------------------------------------------------
__global__ void k_f32_to_bf16(const float* __restrict__ src, __bf16* __restrict__ dst, int n) {
    int i = blockIdx.x * blockDim.x + threadIdx.x;
    int stride = gridDim.x * blockDim.x;
    for (; i < n; i += stride) dst[i] = (__bf16)src[i];
}

// ---------------------------------------------------------------------------
// init: h (bf16), c (f32), symbol = SOS, emit symbol at t=0
// ---------------------------------------------------------------------------
__global__ void k_init(const float* __restrict__ h0, const float* __restrict__ c0,
                       __bf16* __restrict__ h_bf, float* __restrict__ c,
                       int* __restrict__ sym, float* __restrict__ out_sym) {
    int b = blockIdx.x;
    for (int k = threadIdx.x; k < H_; k += blockDim.x) {
        h_bf[b * H_ + k] = (__bf16)h0[b * H_ + k];
        c[b * H_ + k] = c0[b * H_ + k];
    }
    if (threadIdx.x == 0) { sym[b] = SOS; out_sym[(size_t)b * T_] = (float)SOS; }
}

// ---------------------------------------------------------------------------
// t=0 logits: dec0 = embed[SOS] @ w_out^T + b_out, identical for all batches.
// Exact f32 (one-time cost, ~131MB read).
// ---------------------------------------------------------------------------
__global__ void k_dec0(const float* __restrict__ embed, const float* __restrict__ w_out,
                       const float* __restrict__ b_out, float* __restrict__ dec_out) {
    int v = blockIdx.x * blockDim.x + threadIdx.x;
    if (v >= V_) return;
    const float* e = embed + (size_t)SOS * H_;
    const float* w = w_out + (size_t)v * H_;
    float s = b_out[v];
    for (int k = 0; k < H_; ++k) s = fmaf(e[k], w[k], s);
    for (int b = 0; b < B_; ++b) dec_out[(size_t)b * T_ * V_ + v] = s;
}

// ---------------------------------------------------------------------------
// gather current symbols' embedding rows -> bf16 activations
// ---------------------------------------------------------------------------
__global__ void k_gather_x(const float* __restrict__ embed, const int* __restrict__ sym,
                           __bf16* __restrict__ x_bf) {
    int b = blockIdx.x;
    const float* e = embed + (size_t)sym[b] * H_;
    for (int k = threadIdx.x; k < H_; k += blockDim.x) x_bf[b * H_ + k] = (__bf16)e[k];
}

// ---------------------------------------------------------------------------
// Per-wave GEMM core: 2 N-tiles x 4 M-tiles (all 64 batch rows) so each
// weight (B) fragment is loaded from L2 exactly ONCE per step; the tiny A
// matrix (64x1024 bf16 = 128KB, shared by all waves of a block) is re-read
// out of WGP$/L1.  v_wmma_f32_16x16x32_bf16, fragment layouts used by this code (same k order in A and B):
//   A 16x32 bf16: lane l -> row (l&15); K-half 16*(l>>4); 16 contig K
//   B 32x16 bf16 (= W^T): lane l -> col (l&15); K-half 16*(l>>4); 16 contig K
// ---------------------------------------------------------------------------
__device__ __forceinline__ void wave_gemm_m4n2(const __bf16* __restrict__ A,  // [64][K]
                                               const __bf16* __restrict__ W,  // [N][K]
                                               int K, int ncol0, v8f acc[8]) {
    int lane = threadIdx.x & 31;
    int r  = lane & 15;
    int hi = lane >> 4;
    const __bf16* w0 = W + (size_t)(ncol0 + r) * K + hi * 16;
    const __bf16* w1 = W + (size_t)(ncol0 + 16 + r) * K + hi * 16;
    const __bf16* a0 = A + (size_t)r * K + hi * 16;
    for (int kb = 0; kb < K; kb += 32) {
        AFrag b0, b1;
        b0.h[0] = *reinterpret_cast<const bf16x8*>(w0 + kb);
        b0.h[1] = *reinterpret_cast<const bf16x8*>(w0 + kb + 8);
        b1.h[0] = *reinterpret_cast<const bf16x8*>(w1 + kb);
        b1.h[1] = *reinterpret_cast<const bf16x8*>(w1 + kb + 8);
        #pragma unroll
        for (int m = 0; m < 4; ++m) {
            AFrag a;
            const __bf16* am = a0 + (size_t)(16 * m) * K;
            a.h[0] = *reinterpret_cast<const bf16x8*>(am + kb);
            a.h[1] = *reinterpret_cast<const bf16x8*>(am + kb + 8);
            acc[m]     = __builtin_amdgcn_wmma_f32_16x16x32_bf16(false, a.v, false, b0.v,
                                                                 (short)0, acc[m], false, false);
            acc[m + 4] = __builtin_amdgcn_wmma_f32_16x16x32_bf16(false, a.v, false, b1.v,
                                                                 (short)0, acc[m + 4], false, false);
        }
    }
}

// ---------------------------------------------------------------------------
// gates = x@w_ih^T + h@w_hh^T + b_ih + b_hh   [64 x 4096]
// grid 16 blocks, block 256 = 8 waves, wave -> 2 N-tiles x all-M
// w_ih/w_hh each streamed exactly once per step (16.8MB bf16 total).
// ---------------------------------------------------------------------------
__global__ void k_gates(const __bf16* __restrict__ x_bf, const __bf16* __restrict__ h_bf,
                        const __bf16* __restrict__ w_ih_bf, const __bf16* __restrict__ w_hh_bf,
                        const float* __restrict__ b_ih, const float* __restrict__ b_hh,
                        float* __restrict__ gates) {
    int wave = threadIdx.x >> 5;
    int lane = threadIdx.x & 31;
    int ncol0 = (blockIdx.x * 8 + wave) * 32;
    v8f acc[8] = {};
    wave_gemm_m4n2(x_bf, w_ih_bf, H_, ncol0, acc);
    wave_gemm_m4n2(h_bf, w_hh_bf, H_, ncol0, acc);
    int hi = lane >> 4, col = lane & 15;
    int n0 = ncol0 + col, n1 = n0 + 16;
    float bias0 = b_ih[n0] + b_hh[n0];
    float bias1 = b_ih[n1] + b_hh[n1];
    #pragma unroll
    for (int m = 0; m < 4; ++m) {
        #pragma unroll
        for (int r = 0; r < 8; ++r) {
            int row = 16 * m + r + 8 * hi;
            gates[(size_t)row * G4H + n0] = acc[m][r] + bias0;
            gates[(size_t)row * G4H + n1] = acc[m + 4][r] + bias1;
        }
    }
}

// ---------------------------------------------------------------------------
// LSTM pointwise: c2 = sig(f)*c + sig(i)*tanh(g); h2 = sig(o)*tanh(c2)
// h kept in bf16 (only consumed by the bf16 WMMA GEMMs); c stays f32.
// ---------------------------------------------------------------------------
__global__ void k_lstm_pw(const float* __restrict__ gates, float* __restrict__ c,
                          __bf16* __restrict__ h_bf) {
    int b = blockIdx.x;
    const float* g = gates + (size_t)b * G4H;
    for (int k = threadIdx.x; k < H_; k += blockDim.x) {
        float i  = g[k];
        float f  = g[H_ + k];
        float gg = g[2 * H_ + k];
        float o  = g[3 * H_ + k];
        float c2 = sigmoidf_(f) * c[b * H_ + k] + sigmoidf_(i) * tanhf(gg);
        float h2 = sigmoidf_(o) * tanhf(c2);
        c[b * H_ + k] = c2;
        h_bf[b * H_ + k] = (__bf16)h2;
    }
}

// ---------------------------------------------------------------------------
// dec = h@w_out^T + b_out  [64 x 32000] -> d_out[:,t,:]
// grid 125 blocks, block 256 = 8 waves, wave -> 2 N-tiles x all-M.
// Dominant kernel: w_out bf16 (65.5MB) streamed from L2 exactly once/step.
// ---------------------------------------------------------------------------
__global__ void k_outgemm(const __bf16* __restrict__ h_bf, const __bf16* __restrict__ w_out_bf,
                          const float* __restrict__ b_out, float* __restrict__ dec_out, int t) {
    int wave = threadIdx.x >> 5;
    int lane = threadIdx.x & 31;
    int ncol0 = (blockIdx.x * 8 + wave) * 32;
    v8f acc[8] = {};
    wave_gemm_m4n2(h_bf, w_out_bf, H_, ncol0, acc);
    int hi = lane >> 4, col = lane & 15;
    int n0 = ncol0 + col, n1 = n0 + 16;
    float bias0 = b_out[n0];
    float bias1 = b_out[n1];
    #pragma unroll
    for (int m = 0; m < 4; ++m) {
        #pragma unroll
        for (int r = 0; r < 8; ++r) {
            int b = 16 * m + r + 8 * hi;
            size_t base = ((size_t)b * T_ + t) * V_;
            dec_out[base + n0] = acc[m][r] + bias0;
            dec_out[base + n1] = acc[m + 4][r] + bias1;
        }
    }
}

// ---------------------------------------------------------------------------
// argmax over V per batch row (first-occurrence tie-break like jnp.argmax)
// ---------------------------------------------------------------------------
__global__ void k_argmax(const float* __restrict__ dec_out, int t,
                         int* __restrict__ sym, float* __restrict__ out_sym) {
    __shared__ float sv[256];
    __shared__ int   si[256];
    int b = blockIdx.x;
    const float* row = dec_out + ((size_t)b * T_ + t) * V_;
    float best = -3.402823466e38f;
    int bi = 0;
    for (int v = threadIdx.x; v < V_; v += blockDim.x) {
        float x = row[v];
        if (x > best) { best = x; bi = v; }
    }
    sv[threadIdx.x] = best; si[threadIdx.x] = bi;
    __syncthreads();
    for (int s = 128; s > 0; s >>= 1) {
        if ((int)threadIdx.x < s) {
            float ov = sv[threadIdx.x + s]; int oi = si[threadIdx.x + s];
            if (ov > sv[threadIdx.x] || (ov == sv[threadIdx.x] && oi < si[threadIdx.x])) {
                sv[threadIdx.x] = ov; si[threadIdx.x] = oi;
            }
        }
        __syncthreads();
    }
    if (threadIdx.x == 0) {
        sym[b] = si[0];
        out_sym[(size_t)b * T_ + t] = (float)si[0];
    }
}

// ---------------------------------------------------------------------------
extern "C" void kernel_launch(void* const* d_in, const int* in_sizes, int n_in,
                              void* d_out, int out_size, void* d_ws, size_t ws_size,
                              hipStream_t stream) {
    // setup_inputs order:
    // 0 output, 1 h0, 2 c0, 3 target_outputs, 4 target_lengths,
    // 5 embed, 6 w_ih, 7 w_hh, 8 b_ih, 9 b_hh, 10 w_out, 11 b_out
    const float* h0    = (const float*)d_in[1];
    const float* c0    = (const float*)d_in[2];
    const float* embed = (const float*)d_in[5];
    const float* w_ih  = (const float*)d_in[6];
    const float* w_hh  = (const float*)d_in[7];
    const float* b_ih  = (const float*)d_in[8];
    const float* b_hh  = (const float*)d_in[9];
    const float* w_out = (const float*)d_in[10];
    const float* b_out = (const float*)d_in[11];
    (void)in_sizes; (void)n_in; (void)out_size; (void)ws_size;

    // Workspace layout (~83.9 MB)
    char* ws = (char*)d_ws;
    __bf16* w_ih_bf  = (__bf16*)(ws + 0);                    //  8,388,608
    __bf16* w_hh_bf  = (__bf16*)(ws + 8388608);              //  8,388,608
    __bf16* w_out_bf = (__bf16*)(ws + 16777216);             // 65,536,000
    __bf16* h_bf     = (__bf16*)(ws + 82313216);             //    131,072
    __bf16* x_bf     = (__bf16*)(ws + 82444288);             //    131,072
    float*  cbuf     = (float*) (ws + 82575360);             //    262,144
    float*  gates    = (float*) (ws + 82837504);             //  1,048,576
    int*    sym      = (int*)   (ws + 83886080);             //        256

    float* out     = (float*)d_out;                          // [B, T, V] logits
    float* out_sym = out + (size_t)B_ * T_ * V_;             // [B, T] symbols (as float)

    // One-time: bf16 weight conversion, state init, t=0 logits
    k_f32_to_bf16<<<1024, 256, 0, stream>>>(w_ih,  w_ih_bf,  G4H * H_);
    k_f32_to_bf16<<<1024, 256, 0, stream>>>(w_hh,  w_hh_bf,  G4H * H_);
    k_f32_to_bf16<<<4096, 256, 0, stream>>>(w_out, w_out_bf, V_ * H_);
    k_init<<<B_, 256, 0, stream>>>(h0, c0, h_bf, cbuf, sym, out_sym);
    k_dec0<<<(V_ + 255) / 256, 256, 0, stream>>>(embed, w_out, b_out, out);

    // Sequential decode: emit (sym_t, dec_t) for t = 1..T-1
    for (int t = 1; t < T_; ++t) {
        k_gather_x<<<B_, 256, 0, stream>>>(embed, sym, x_bf);
        k_gates<<<G4H / 256, 256, 0, stream>>>(x_bf, h_bf, w_ih_bf, w_hh_bf,
                                               b_ih, b_hh, gates);
        k_lstm_pw<<<B_, 256, 0, stream>>>(gates, cbuf, h_bf);
        k_outgemm<<<V_ / 256, 256, 0, stream>>>(h_bf, w_out_bf, b_out, out, t);
        k_argmax<<<B_, 256, 0, stream>>>(out, t, sym, out_sym);
    }
}
